// MambaStateEstimator_14998025797893
// MI455X (gfx1250) — matrix operand order repaired and checked by reading, not compiled, not run on hardware
//
#include <hip/hip_runtime.h>
#include <hip/hip_bf16.h>
#include <math.h>

// ---------------------------------------------------------------------------
// Problem constants (from reference)
// ---------------------------------------------------------------------------
#define BATCH    2
#define TSEQ     1024
#define DMODEL   1024
#define DINNER   2048
#define DSTATE   16
#define DCONV    4
#define DTRANK   64
#define NLAYERS  4
#define NOBS     64
#define NSTATE   32
#define ROWS     (BATCH * TSEQ)       // 2048 tokens

typedef __attribute__((ext_vector_type(16))) __bf16 v16bf;
typedef __attribute__((ext_vector_type(8)))  __bf16 v8bf;
typedef __attribute__((ext_vector_type(4)))  __bf16 v4bf;
typedef __attribute__((ext_vector_type(2)))  __bf16 v2bf;
typedef __attribute__((ext_vector_type(8)))  float  v8f;
typedef __attribute__((ext_vector_type(4)))  float  v4f;   // builtin vector:
                                                           // copyable across
                                                           // address spaces

// Explicit global-address-space pointers so staging loads lower to
// global_load_* (LOADcnt only) instead of flat_load_* (LOADcnt + DScnt,
// ordered with the LDS pipe).
typedef const __attribute__((address_space(1))) float* gfp_t;
typedef const __attribute__((address_space(1))) v4f*   gv4p_t;

// Activation codes for the GEMM epilogue
#define ACT_NONE     0
#define ACT_GELU     1
#define ACT_SOFTPLUS 2

// LDS row pitch in bf16 elements: 64 K-elems + 8 pad.
// Pitch = 144 bytes = 36 dwords -> 16 rows tile all 64 banks (gcd(36,64)=4),
// and 144 is a multiple of 16 so every b128 fragment load stays aligned.
#define APITCH 72

// ---------------------------------------------------------------------------
// WMMA GEMM:  C[M,N] = act(A[M,K] * B[K,N] + bias) (+ residual)
//   - fp32 inputs converted to bf16 in LDS, fp32 accumulate via
//     v_wmma_f32_16x16x32_bf16 (wave32, 16x16x32).
//   - block = 256 threads = 8 waves; block tile 64(M) x 32*NT(N); K step 64.
//   - NT = n-tiles per wave (1 or 2). NT=2 reuses each A fragment for two
//     WMMAs: 4 WMMAs per 12 b128 LDS loads per K-iteration.
//   - A tile row-major (M x K): a lane's 8 contiguous K values are one
//     ds_load_b128. B tile TRANSPOSED (N x K): a B fragment's two runs of 8
//     contiguous K values are two ds_load_b128.
//   - Staging pointers are explicit AS(1) (global_load_*), precomputed and
//     advanced additively; loads batched before converts/stores.
//   Requirements: M%64==0, N%(32*NT)==0, K%64==0, lda%4==0.
// ---------------------------------------------------------------------------
template<int NT>
__global__ __launch_bounds__(256)
void gemm_bf16_wmma(const float* __restrict__ A, int lda,
                    const float* __restrict__ Bm, int ldb,
                    float* __restrict__ C, int ldc,
                    const float* __restrict__ bias,
                    const float* __restrict__ residual, int ldr,
                    int K, int act)
{
    constexpr int NB    = 32 * NT;              // block tile N
    constexpr int LOGNB = (NT == 2) ? 6 : 5;
    constexpr int BPT   = NB / 8;               // B K-pairs per thread (4*NT)

    __shared__ __bf16 As [64][APITCH];          // M x K
    __shared__ __bf16 BsT[NB][APITCH];          // N x K (transposed)

    const int tid  = threadIdx.x;
    const int wave = tid >> 5;
    const int lane = tid & 31;
    const int half = lane >> 4;                 // 0: lanes 0-15, 1: 16-31
    const int l16  = lane & 15;

    const int am  = (wave & 3) * 16;            // wave M offset in block tile
    const int bn  = (wave >> 2) * (16 * NT);    // wave N offset in block tile
    const int gm0 = blockIdx.y * 64;
    const int gn0 = blockIdx.x * NB;

    const gfp_t Ag = (gfp_t)A;
    const gfp_t Bg = (gfp_t)Bm;

    // ---- precompute staging pointers (advanced additively per K step) ----
    gfp_t   ag[4];                              // A: 4 vec4 chunks / thread
    __bf16* al[4];
    #pragma unroll
    for (int i = 0; i < 4; ++i) {
        const int cix = i * 256 + tid;
        const int r   = cix >> 4;               // 16 chunks per 64-col row
        const int c   = (cix & 15) * 4;
        ag[i] = Ag + (size_t)(gm0 + r) * lda + c;
        al[i] = &As[r][c];
    }
    gfp_t   bg[BPT];                            // B: K-pairs, transposed store
    __bf16* bl[BPT];
    #pragma unroll
    for (int j = 0; j < BPT; ++j) {
        const int pix = j * 256 + tid;
        const int n   = pix & (NB - 1);
        const int k2  = pix >> LOGNB;           // K-pair index 0..31
        bg[j] = Bg + (size_t)(2 * k2) * ldb + gn0 + n;
        bl[j] = &BsT[n][2 * k2];
    }
    const size_t bstep = (size_t)64 * ldb;

    // generic-pointer mirrors for __builtin_prefetch
    const float* apf = A  + (size_t)(gm0 + (tid >> 5)) * lda + (tid & 31);
    const float* bpf = Bm + (size_t)(tid >> 5) * ldb + gn0 + (tid & 31);

    v8f acc[NT];
    #pragma unroll
    for (int t = 0; t < NT; ++t) acc[t] = (v8f){};

    for (int k0 = 0; k0 < K; k0 += 64) {
        // ---- batched global loads (overlapping loadcnt) ----
        v4f fa[4];
        #pragma unroll
        for (int i = 0; i < 4; ++i) fa[i] = *(gv4p_t)ag[i];
        float fb0[BPT], fb1[BPT];
        #pragma unroll
        for (int j = 0; j < BPT; ++j) { fb0[j] = bg[j][0]; fb1[j] = bg[j][ldb]; }

        // ---- convert + packed LDS stores ----
        #pragma unroll
        for (int i = 0; i < 4; ++i) {
            v4bf p;
            p[0] = (__bf16)fa[i][0]; p[1] = (__bf16)fa[i][1];
            p[2] = (__bf16)fa[i][2]; p[3] = (__bf16)fa[i][3];
            *(v4bf*)al[i] = p;                  // ds_store_b64
        }
        #pragma unroll
        for (int j = 0; j < BPT; ++j) {
            v2bf p;
            p[0] = (__bf16)fb0[j]; p[1] = (__bf16)fb1[j];
            *(v2bf*)bl[j] = p;                  // ds_store_b32
        }

        // ---- advance pointers; prefetch next K tile ----
        #pragma unroll
        for (int i = 0; i < 4; ++i) ag[i] += 64;
        #pragma unroll
        for (int j = 0; j < BPT; ++j) bg[j] += bstep;
        apf += 64; bpf += bstep;
        if (k0 + 64 < K) {
            __builtin_prefetch(apf, 0, 3);
            __builtin_prefetch(bpf, 0, 3);
        }
        __syncthreads();

        // ---- 2 K-halves x NT n-tiles of WMMA; b128 fragment loads ----
        #pragma unroll
        for (int kk = 0; kk < 2; ++kk) {
            const int ko = kk * 32;
            // A fragment (16x32): row m = l16 (both halves); half selects K:
            // e[0..7] -> K = ko+half*8+j, e[8..15] -> +16.
            union { v16bf v; v8bf h[2]; } au;
            au.h[0] = *(const v8bf*)&As[am + l16][ko + half * 8];
            au.h[1] = *(const v8bf*)&As[am + l16][ko + 16 + half * 8];
            #pragma unroll
            for (int t = 0; t < NT; ++t) {
                // B fragment (32x16): col n = l16; e[0..7] -> K = ko+half*8+j,
                // e[8..15] -> +16 (same order as A).
                union { v16bf v; v8bf h[2]; } bu;
                bu.h[0] = *(const v8bf*)&BsT[bn + t * 16 + l16][ko + half * 8];
                bu.h[1] = *(const v8bf*)&BsT[bn + t * 16 + l16][ko + 16 + half * 8];
                acc[t] = __builtin_amdgcn_wmma_f32_16x16x32_bf16(
                             false, au.v, false, bu.v, (short)0, acc[t],
                             false, false);
            }
        }
        __syncthreads();
    }

    // ---- epilogue: C/D layout: VGPR r -> row M = half*8+r, col N = lane%16
    #pragma unroll
    for (int t = 0; t < NT; ++t) {
        const int gn = gn0 + bn + t * 16 + l16;
        const float bv = bias ? bias[gn] : 0.0f;
        #pragma unroll
        for (int r = 0; r < 8; ++r) {
            const int gm = gm0 + am + half * 8 + r;
            float v = acc[t][r] + bv;
            if (act == ACT_GELU) {
                v = 0.5f * v * (1.0f + erff(v * 0.70710678118654752f));
            } else if (act == ACT_SOFTPLUS) {
                v = (v > 20.0f) ? v : log1pf(expf(v));
            }
            if (residual) v += residual[(size_t)gm * ldr + gn];
            C[(size_t)gm * ldc + gn] = v;
        }
    }
}

// ---------------------------------------------------------------------------
// LayerNorm: one block (256 threads) per row of D elements.
// ---------------------------------------------------------------------------
__global__ __launch_bounds__(256)
void ln_kernel(const float* __restrict__ x,
               const float* __restrict__ g,
               const float* __restrict__ b,
               float* __restrict__ out, int D)
{
    __shared__ float red[256];
    const int tid = threadIdx.x;
    const int row = blockIdx.x;
    const float* xr = x + (size_t)row * D;

    float s = 0.0f;
    for (int i = tid; i < D; i += 256) s += xr[i];
    red[tid] = s; __syncthreads();
    for (int off = 128; off > 0; off >>= 1) {
        if (tid < off) red[tid] += red[tid + off];
        __syncthreads();
    }
    const float mean = red[0] / (float)D;
    __syncthreads();

    float vs = 0.0f;
    for (int i = tid; i < D; i += 256) { float d = xr[i] - mean; vs += d * d; }
    red[tid] = vs; __syncthreads();
    for (int off = 128; off > 0; off >>= 1) {
        if (tid < off) red[tid] += red[tid + off];
        __syncthreads();
    }
    const float inv = rsqrtf(red[0] / (float)D + 1e-5f);

    float* orow = out + (size_t)row * D;
    for (int i = tid; i < D; i += 256)
        orow[i] = (xr[i] - mean) * inv * g[i] + b[i];
}

// ---------------------------------------------------------------------------
// Causal depthwise conv (width 4) over time + bias + SiLU.
// u = first DINNER columns of xz (row stride 2*DINNER).
// ---------------------------------------------------------------------------
__global__ __launch_bounds__(256)
void conv_silu_kernel(const float* __restrict__ xz,
                      const float* __restrict__ cw,   // [DINNER][4]
                      const float* __restrict__ cb,   // [DINNER]
                      float* __restrict__ uc)         // [ROWS][DINNER]
{
    const int idx = blockIdx.x * 256 + threadIdx.x;
    if (idx >= ROWS * DINNER) return;
    const int d   = idx % DINNER;
    const int row = idx / DINNER;           // row = b*T + t
    const int t   = row % TSEQ;

    float acc = cb[d];
    #pragma unroll
    for (int k = 0; k < DCONV; ++k) {
        const int tt = t - (DCONV - 1) + k;
        if (tt >= 0)
            acc += xz[(size_t)(row - (DCONV - 1) + k) * (2 * DINNER) + d] * cw[d * DCONV + k];
    }
    uc[idx] = acc / (1.0f + expf(-acc));    // SiLU
}

// ---------------------------------------------------------------------------
// Selective scan: one thread per (batch, channel). 16 states in registers,
// sequential loop over T. Latency-bound by construction (true recurrence).
// dbc row layout: [0:64)=dt_rank proj, [64:80)=B_t, [80:96)=C_t.
// B_t/C_t rows are 16-byte aligned -> v4f vector loads.
// ---------------------------------------------------------------------------
__global__ __launch_bounds__(256)
void scan_kernel(const float* __restrict__ dt,    // [ROWS][DINNER] softplus'd
                 const float* __restrict__ dbc,   // [ROWS][96]
                 const float* __restrict__ uc,    // [ROWS][DINNER]
                 const float* __restrict__ Alog,  // [DINNER][16]
                 float* __restrict__ y)           // [ROWS][DINNER]
{
    const int idx = blockIdx.x * 256 + threadIdx.x;
    if (idx >= BATCH * DINNER) return;
    const int b = idx / DINNER;
    const int d = idx % DINNER;

    float Av[DSTATE], hc[DSTATE];
    #pragma unroll
    for (int s = 0; s < DSTATE; ++s) {
        Av[s] = -expf(Alog[d * DSTATE + s]);
        hc[s] = 0.0f;
    }

    for (int t = 0; t < TSEQ; ++t) {
        const int row = b * TSEQ + t;
        const float dtv = dt[(size_t)row * DINNER + d];
        const float uv  = uc[(size_t)row * DINNER + d];
        const float du  = dtv * uv;
        const v4f* pB = (const v4f*)(dbc + (size_t)row * (DTRANK + 2 * DSTATE) + DTRANK);
        const v4f* pC = pB + (DSTATE / 4);
        float yv = 0.0f;
        #pragma unroll
        for (int i = 0; i < DSTATE / 4; ++i) {
            const v4f bq = pB[i];
            const v4f cq = pC[i];
            #pragma unroll
            for (int q = 0; q < 4; ++q) {
                const int s = 4 * i + q;
                const float dA = expf(dtv * Av[s]);
                hc[s] = dA * hc[s] + du * bq[q];
                yv += hc[s] * cq[q];
            }
        }
        y[(size_t)row * DINNER + d] = yv;
    }
}

// ---------------------------------------------------------------------------
// Gate: y = (y + uc * D_skip) * silu(z), z = columns [DINNER, 2*DINNER) of xz
// ---------------------------------------------------------------------------
__global__ __launch_bounds__(256)
void gate_kernel(float* __restrict__ y,
                 const float* __restrict__ uc,
                 const float* __restrict__ xz,
                 const float* __restrict__ Dp)
{
    const int idx = blockIdx.x * 256 + threadIdx.x;
    if (idx >= ROWS * DINNER) return;
    const int d   = idx % DINNER;
    const int row = idx / DINNER;
    const float z = xz[(size_t)row * (2 * DINNER) + DINNER + d];
    const float sz = z / (1.0f + expf(-z));
    y[idx] = (y[idx] + uc[idx] * Dp[d]) * sz;
}

// ---------------------------------------------------------------------------
// Host orchestration (graph-capture safe: only kernel launches on `stream`)
// ---------------------------------------------------------------------------
static inline void launch_gemm(const float* A, int lda, const float* B, int ldb,
                               float* C, int ldc, const float* bias,
                               const float* residual, int ldr,
                               int M, int N, int K, int act, hipStream_t stream)
{
    if ((N & 63) == 0) {
        gemm_bf16_wmma<2><<<dim3(N / 64, M / 64), 256, 0, stream>>>(
            A, lda, B, ldb, C, ldc, bias, residual, ldr, K, act);
    } else {
        gemm_bf16_wmma<1><<<dim3(N / 32, M / 64), 256, 0, stream>>>(
            A, lda, B, ldb, C, ldc, bias, residual, ldr, K, act);
    }
}

extern "C" void kernel_launch(void* const* d_in, const int* in_sizes, int n_in,
                              void* d_out, int out_size, void* d_ws, size_t ws_size,
                              hipStream_t stream)
{
    (void)in_sizes; (void)n_in; (void)out_size; (void)ws_size;

    const float* Y        = (const float*)d_in[0];
    const float* W_embed  = (const float*)d_in[1];
    const float* b_embed  = (const float*)d_in[2];
    const float* norm_g   = (const float*)d_in[3];
    const float* norm_b   = (const float*)d_in[4];
    const float* W_in     = (const float*)d_in[5];
    const float* conv_w   = (const float*)d_in[6];
    const float* conv_b   = (const float*)d_in[7];
    const float* W_x      = (const float*)d_in[8];
    const float* W_dt     = (const float*)d_in[9];
    const float* b_dt     = (const float*)d_in[10];
    const float* A_log    = (const float*)d_in[11];
    const float* D_skip   = (const float*)d_in[12];
    const float* W_out    = (const float*)d_in[13];
    const float* dec_ln_g = (const float*)d_in[14];
    const float* dec_ln_b = (const float*)d_in[15];
    const float* W1       = (const float*)d_in[16];
    const float* b1       = (const float*)d_in[17];
    const float* W2       = (const float*)d_in[18];
    const float* b2       = (const float*)d_in[19];

    // workspace carve-out (floats)
    float* ws = (float*)d_ws;
    size_t off = 0;
    auto carve = [&](size_t n) { float* p = ws + off; off += n; return p; };
    float* xA   = carve((size_t)ROWS * DMODEL);       // residual stream ping
    float* xB   = carve((size_t)ROWS * DMODEL);       // residual stream pong
    float* hbuf = carve((size_t)ROWS * DMODEL);       // LN output
    float* xz   = carve((size_t)ROWS * 2 * DINNER);   // W_in output
    float* ucb  = carve((size_t)ROWS * DINNER);       // conv+silu output
    float* dbc  = carve((size_t)ROWS * (DTRANK + 2 * DSTATE));
    float* dtb  = carve((size_t)ROWS * DINNER);       // softplus(dt)
    float* yb   = carve((size_t)ROWS * DINNER);       // scan output
    float* h1   = carve((size_t)ROWS * (DMODEL / 2)); // head hidden

    const int ew = ROWS * DINNER;                     // elementwise work items
    const dim3 ewGrid((ew + 255) / 256);

    // 1) embedding: x = Y @ W_embed + b_embed   [2048 x 64] @ [64 x 1024]
    launch_gemm(Y, NOBS, W_embed, DMODEL, xA, DMODEL,
                b_embed, nullptr, 0, ROWS, DMODEL, NOBS, ACT_NONE, stream);

    float* cur = xA;
    float* nxt = xB;
    for (int l = 0; l < NLAYERS; ++l) {
        const float* g   = norm_g + (size_t)l * DMODEL;
        const float* bb  = norm_b + (size_t)l * DMODEL;
        const float* Wi  = W_in   + (size_t)l * DMODEL * 2 * DINNER;
        const float* cw  = conv_w + (size_t)l * DINNER * DCONV;
        const float* cb  = conv_b + (size_t)l * DINNER;
        const float* Wx  = W_x    + (size_t)l * DINNER * (DTRANK + 2 * DSTATE);
        const float* Wdt = W_dt   + (size_t)l * DTRANK * DINNER;
        const float* bdt = b_dt   + (size_t)l * DINNER;
        const float* Al  = A_log  + (size_t)l * DINNER * DSTATE;
        const float* Dp  = D_skip + (size_t)l * DINNER;
        const float* Wo  = W_out  + (size_t)l * DINNER * DMODEL;

        // 2) h = LN(x)
        ln_kernel<<<ROWS, 256, 0, stream>>>(cur, g, bb, hbuf, DMODEL);

        // 3) xz = h @ W_in                      [2048 x 1024] @ [1024 x 4096]
        launch_gemm(hbuf, DMODEL, Wi, 2 * DINNER, xz, 2 * DINNER,
                    nullptr, nullptr, 0, ROWS, 2 * DINNER, DMODEL, ACT_NONE, stream);

        // 4) uc = silu(causal_conv(u) + cb)
        conv_silu_kernel<<<ewGrid, 256, 0, stream>>>(xz, cw, cb, ucb);

        // 5) dbc = uc @ W_x                     [2048 x 2048] @ [2048 x 96]
        launch_gemm(ucb, DINNER, Wx, DTRANK + 2 * DSTATE, dbc, DTRANK + 2 * DSTATE,
                    nullptr, nullptr, 0, ROWS, DTRANK + 2 * DSTATE, DINNER, ACT_NONE, stream);

        // 6) dt = softplus(dbc[:, :64] @ W_dt + b_dt)   (A has lda = 96)
        launch_gemm(dbc, DTRANK + 2 * DSTATE, Wdt, DINNER, dtb, DINNER,
                    bdt, nullptr, 0, ROWS, DINNER, DTRANK, ACT_SOFTPLUS, stream);

        // 7) selective scan (recurrent; states in registers)
        scan_kernel<<<(BATCH * DINNER + 255) / 256, 256, 0, stream>>>(
            dtb, dbc, ucb, Al, yb);

        // 8) y = (y + uc * D_skip) * silu(z)
        gate_kernel<<<ewGrid, 256, 0, stream>>>(yb, ucb, xz, Dp);

        // 9) x = x + y @ W_out                  [2048 x 2048] @ [2048 x 1024]
        launch_gemm(yb, DINNER, Wo, DMODEL, nxt, DMODEL,
                    nullptr, cur, DMODEL, ROWS, DMODEL, DINNER, ACT_NONE, stream);

        float* tmp = cur; cur = nxt; nxt = tmp;
    }

    // 10) head: LN -> GELU(W1) -> W2
    ln_kernel<<<ROWS, 256, 0, stream>>>(cur, dec_ln_g, dec_ln_b, hbuf, DMODEL);

    launch_gemm(hbuf, DMODEL, W1, DMODEL / 2, h1, DMODEL / 2,
                b1, nullptr, 0, ROWS, DMODEL / 2, DMODEL, ACT_GELU, stream);

    launch_gemm(h1, DMODEL / 2, W2, NSTATE, (float*)d_out, NSTATE,
                b2, nullptr, 0, ROWS, NSTATE, DMODEL / 2, ACT_NONE, stream);
}
